// AttentionPooling_39874476376512
// MI455X (gfx1250) — matrix operand order repaired and checked by reading, not compiled, not run on hardware
//
#include <hip/hip_runtime.h>
#include <math.h>
#include <stdint.h>

// ---------------- problem constants ----------------
#define B_      16
#define S_      4096
#define D_      1536
#define H_      8
#define HD_     192
#define SCALE_  0.07216878364870322f   // 1/sqrt(192)
#define CHUNKS_ 16                      // seq chunks per batch
#define TOKC_   256                     // tokens per chunk
#define TILES_  16                      // 16-token tiles per chunk
#define KK_     48                      // 1536 / 32 k-chunks
#define EPS_    1e-6f

// ---------------- workspace layout (float offsets) ----------------
#define OFF_Q     0                         // q[1536]
#define OFF_QKS   1536                      // scale*qk [H][D]
#define OFF_FRAG  13824                     // packed bf16 B-frags [48][32][8] dwords
#define OFF_SB    26112                     // score bias per head [8]
#define OFF_Z     26120                     // Z partials [B][CHUNKS][H]
#define OFF_T     28168                     // t = u/Z  [B][H][D]
#define OFF_POOL  224776                    // pooled [B][D]
#define OFF_Y     249352                    // y = pooled@wo^T+bo [B][D]
#define OFF_U     273928                    // u partials [B][CHUNKS][H][D]
// total = 273928 + 3145728 floats = ~13.7 MB workspace required

typedef __attribute__((ext_vector_type(16))) __bf16 v16bf;
typedef __attribute__((ext_vector_type(8)))  float  v8f;

__device__ __forceinline__ unsigned int pack_bf16(float a, float b) {
  unsigned short lo = __builtin_bit_cast(unsigned short, (__bf16)a);
  unsigned short hi = __builtin_bit_cast(unsigned short, (__bf16)b);
  return ((unsigned int)hi << 16) | lo;
}

// K0: q[e] = query . wq[e,:] + bq[e]   (one block per e, coalesced row reduce)
__global__ void k_proj_q(const float* __restrict__ query, const float* __restrict__ wq,
                         const float* __restrict__ bq, float* __restrict__ ws) {
  __shared__ float red[256];
  int e = blockIdx.x, t = threadIdx.x;
  const float* wr = wq + (size_t)e * D_;
  float acc = 0.f;
  for (int i = 0; i < 6; ++i) { int d = t + i * 256; acc += query[d] * wr[d]; }
  red[t] = acc; __syncthreads();
  for (int off = 128; off > 0; off >>= 1) {
    if (t < off) red[t] += red[t + off];
    __syncthreads();
  }
  if (t == 0) ws[OFF_Q + e] = red[0] + bq[e];
}

// K1: qks[h,d] = scale * sum_j q[h*HD+j] * wk[h*HD+j, d]; also sb[h]
__global__ void k_qk(const float* __restrict__ wk, const float* __restrict__ bk,
                     float* __restrict__ ws) {
  int h = blockIdx.x / 6, dc = blockIdx.x % 6;
  int d = dc * 256 + threadIdx.x;
  const float* q = ws + OFF_Q + h * HD_;
  float acc = 0.f;
  for (int j = 0; j < HD_; ++j) acc += q[j] * wk[(size_t)(h * HD_ + j) * D_ + d];
  ws[OFF_QKS + h * D_ + d] = SCALE_ * acc;
  if (dc == 0 && threadIdx.x == 0) {
    float sb = 0.f;
    for (int j = 0; j < HD_; ++j) sb += q[j] * bk[h * HD_ + j];
    ws[OFF_SB + h] = SCALE_ * sb;
  }
}

// K2: pack qk^T into WMMA bf16 B-matrix fragments (16x16x32, 32x16 B tile).
// B layout: VGPR j (j<4): lanes 0-15 hold K=2j,2j+1 (N=lane), lanes 16-31 K=8+2j,8+2j+1; VGPR j>=4: K+=16-8.
__global__ void k_pack(float* __restrict__ ws) {
  int kk = blockIdx.x, t = threadIdx.x;
  int lane = t >> 3, j = t & 7;
  int n = lane & 15;
  int kb = kk * 32 + ((lane & 16) ? 8 : 0) + 2 * (j & 3) + 16 * (j >> 2);
  float v0 = 0.f, v1 = 0.f;
  if (n < H_) { v0 = ws[OFF_QKS + n * D_ + kb]; v1 = ws[OFF_QKS + n * D_ + kb + 1]; }
  ((unsigned int*)(ws + OFF_FRAG))[kk * 256 + t] = pack_bf16(v0, v1);
}

// K3: main single-pass kernel. One workgroup per (batch, 256-token chunk).
// Wave 0 computes the 16x8 score tile with v_wmma_f32_16x16x32_bf16; all 8
// waves do the weighted accumulation with register-resident u accumulators.
__global__ void __launch_bounds__(256) k_attn(const float* __restrict__ tokens,
                                              const unsigned char* __restrict__ mask,
                                              float* __restrict__ ws) {
  extern __shared__ float smem[];
  float* xt = smem;                                  // 16*1536 f32 tile (96 KB)
  float* wl = smem + 24576;                          // weights wl[h][s] (512 B)
  unsigned int* qb = (unsigned int*)(smem + 24704);  // B-frags (48 KB)

  int tid = threadIdx.x;
  int b = blockIdx.x / CHUNKS_, c = blockIdx.x % CHUNKS_;

  // stage qk B-fragments into LDS (hot for all 48 WMMAs per tile)
  const unsigned int* fragg = (const unsigned int*)(ws + OFF_FRAG);
  for (int i = 0; i < 48; ++i) qb[tid + i * 256] = fragg[tid + i * 256];

  int lane = tid, n = tid & 15, hi = (tid >> 4) & 1;
  float sbh = 0.f;
  if (tid < 32 && n < H_) sbh = ws[OFF_SB + n];

  float acc[48];
#pragma unroll
  for (int i = 0; i < 48; ++i) acc[i] = 0.f;
  float zacc[8];
#pragma unroll
  for (int h = 0; h < 8; ++h) zacc[h] = 0.f;

  const float* xbase = tokens + ((size_t)b * S_ + c * TOKC_) * D_;
  const unsigned long long* mbase =
      (const unsigned long long*)(mask + (size_t)b * S_ + c * TOKC_);
  int d0 = tid * 6;

  for (int tile = 0; tile < TILES_; ++tile) {
    __syncthreads();  // previous tile fully consumed before overwrite
    // coalesced cooperative tile load: 16 tokens x 1536 f32
    const float4* g = (const float4*)(xbase + (size_t)tile * 16 * D_);
    float4* xt4 = (float4*)xt;
#pragma unroll
    for (int i = 0; i < 24; ++i) xt4[tid + i * 256] = g[tid + i * 256];
    __syncthreads();

    if (tid < 32) {  // wave 0: score tile via WMMA
      v8f cf = {};
#pragma unroll 4
      for (int kk = 0; kk < KK_; ++kk) {
        // A-frag (16-bit A 16x32): lane<16 row=n holds K {0..7,16..23}; lane>=16 +8
        const float* xr = xt + n * D_ + kk * 32 + hi * 8;
        float xa[16];
        *(float4*)(xa + 0)  = *(const float4*)(xr);
        *(float4*)(xa + 4)  = *(const float4*)(xr + 4);
        *(float4*)(xa + 8)  = *(const float4*)(xr + 16);
        *(float4*)(xa + 12) = *(const float4*)(xr + 20);
        v16bf a;
#pragma unroll
        for (int i = 0; i < 16; ++i) a[i] = (__bf16)xa[i];
        v16bf bfr = *(const v16bf*)(qb + (kk * 32 + lane) * 8);
        cf = __builtin_amdgcn_wmma_f32_16x16x32_bf16(false, a, false, bfr,
                                                     (short)0, cf, false, false);
      }
      // C layout: VGPR r -> M = r + 8*hi, N = lane&15. Heads are N<8.
      if (n < H_) {
        unsigned long long mb = mbase[tile * 2 + hi];
#pragma unroll
        for (int r = 0; r < 8; ++r) {
          float sc = cf[r] + sbh;
          float e = ((mb >> (r * 8)) & 0xffULL) ? __expf(sc) : 0.f;
          wl[n * 16 + hi * 8 + r] = e;  // wl[h][s]
        }
      }
    }
    __syncthreads();

    if (tile + 1 < TILES_)  // global_prefetch_b8 of next tile
      __builtin_prefetch(xbase + (size_t)(tile + 1) * 16 * D_ + tid * 96, 0, 0);

    // accumulation: thread owns d in [6*tid, 6*tid+6) for all 8 heads
    for (int s = 0; s < 16; ++s) {
      float w[8];
#pragma unroll
      for (int h = 0; h < 8; ++h) { w[h] = wl[h * 16 + s]; zacc[h] += w[h]; }
      const float* xr = xt + s * D_ + d0;
      float xv[6];
      *(float2*)(xv + 0) = *(const float2*)(xr);
      *(float2*)(xv + 2) = *(const float2*)(xr + 2);
      *(float2*)(xv + 4) = *(const float2*)(xr + 4);
#pragma unroll
      for (int h = 0; h < 8; ++h)
#pragma unroll
        for (int j = 0; j < 6; ++j)
          acc[h * 6 + j] = fmaf(w[h], xv[j], acc[h * 6 + j]);
    }
  }

  // write partials
  float* U = ws + OFF_U + ((size_t)b * CHUNKS_ + c) * H_ * D_;
#pragma unroll
  for (int h = 0; h < 8; ++h)
#pragma unroll
    for (int j = 0; j < 6; ++j)
      U[h * D_ + d0 + j] = acc[h * 6 + j];
  if (tid == 0) {
    float* Z = ws + OFF_Z + (b * CHUNKS_ + c) * H_;
#pragma unroll
    for (int h = 0; h < 8; ++h) Z[h] = zacc[h];
  }
}

// K4: t[b,h,d] = sum_c U / sum_c Z
__global__ void k_reduce(float* __restrict__ ws) {
  int gid = blockIdx.x * 256 + threadIdx.x;   // < B*H*D
  int d = gid % D_;
  int h = (gid / D_) % H_;
  int b = gid / (D_ * H_);
  float su = 0.f, sz = 0.f;
  for (int c = 0; c < CHUNKS_; ++c) {
    su += ws[OFF_U + (((size_t)b * CHUNKS_ + c) * H_ + h) * D_ + d];
    sz += ws[OFF_Z + (b * CHUNKS_ + c) * H_ + h];
  }
  ws[OFF_T + ((size_t)b * H_ + h) * D_ + d] = su / sz;
}

// K5: pooled[b, e] = wv[e,:] . t[b, e/HD, :] + bv[e]   (row-per-block, coalesced)
__global__ void k_wv(const float* __restrict__ wv, const float* __restrict__ bv,
                     float* __restrict__ ws) {
  __shared__ float red[16 * 256];
  int e = blockIdx.x, t = threadIdx.x;
  int h = e / HD_;
  const float* wr = wv + (size_t)e * D_;
  const float* tb = ws + OFF_T + h * D_;
  float p[16];
#pragma unroll
  for (int b = 0; b < 16; ++b) p[b] = 0.f;
  for (int i = 0; i < 6; ++i) {
    int d = t + i * 256;
    float wvv = wr[d];
#pragma unroll
    for (int b = 0; b < 16; ++b) p[b] += wvv * tb[(size_t)b * H_ * D_ + d];
  }
#pragma unroll
  for (int b = 0; b < 16; ++b) red[b * 256 + t] = p[b];
  __syncthreads();
  if (t < 16) {
    float s = 0.f;
    for (int i = 0; i < 256; ++i) s += red[t * 256 + i];
    ws[OFF_POOL + t * D_ + e] = s + bv[e];
  }
}

// K6: y[b, e'] = wo[e',:] . pooled[b,:] + bo[e']
__global__ void k_wo(const float* __restrict__ wo, const float* __restrict__ bo,
                     float* __restrict__ ws) {
  __shared__ float red[16 * 256];
  int e = blockIdx.x, t = threadIdx.x;
  const float* wr = wo + (size_t)e * D_;
  const float* pb = ws + OFF_POOL;
  float p[16];
#pragma unroll
  for (int b = 0; b < 16; ++b) p[b] = 0.f;
  for (int i = 0; i < 6; ++i) {
    int d = t + i * 256;
    float wov = wr[d];
#pragma unroll
    for (int b = 0; b < 16; ++b) p[b] += wov * pb[(size_t)b * D_ + d];
  }
#pragma unroll
  for (int b = 0; b < 16; ++b) red[b * 256 + t] = p[b];
  __syncthreads();
  if (t < 16) {
    float s = 0.f;
    for (int i = 0; i < 256; ++i) s += red[t * 256 + i];
    ws[OFF_Y + t * D_ + e] = s + bo[e];
  }
}

// K7: LayerNorm per batch row
__global__ void k_ln(const float* __restrict__ gamma, const float* __restrict__ beta,
                     const float* __restrict__ ws, float* __restrict__ out) {
  __shared__ float red[256];
  __shared__ float s_mu, s_rstd;
  int b = blockIdx.x, t = threadIdx.x;
  const float* y = ws + OFF_Y + (size_t)b * D_;
  float s = 0.f;
  for (int i = 0; i < 6; ++i) s += y[t + i * 256];
  red[t] = s; __syncthreads();
  for (int off = 128; off > 0; off >>= 1) {
    if (t < off) red[t] += red[t + off];
    __syncthreads();
  }
  if (t == 0) s_mu = red[0] / (float)D_;
  __syncthreads();
  float mu = s_mu;
  float v = 0.f;
  for (int i = 0; i < 6; ++i) { float d = y[t + i * 256] - mu; v += d * d; }
  red[t] = v; __syncthreads();
  for (int off = 128; off > 0; off >>= 1) {
    if (t < off) red[t] += red[t + off];
    __syncthreads();
  }
  if (t == 0) s_rstd = rsqrtf(red[0] / (float)D_ + EPS_);
  __syncthreads();
  float rs = s_rstd;
  for (int i = 0; i < 6; ++i) {
    int d = t + i * 256;
    out[(size_t)b * D_ + d] = gamma[d] * (y[d] - mu) * rs + beta[d];
  }
}

extern "C" void kernel_launch(void* const* d_in, const int* in_sizes, int n_in,
                              void* d_out, int out_size, void* d_ws, size_t ws_size,
                              hipStream_t stream) {
  const float* tokens        = (const float*)d_in[0];
  const unsigned char* mask  = (const unsigned char*)d_in[1];
  const float* query         = (const float*)d_in[2];
  const float* wq            = (const float*)d_in[3];
  const float* wk            = (const float*)d_in[4];
  const float* wv            = (const float*)d_in[5];
  const float* bq            = (const float*)d_in[6];
  const float* bk            = (const float*)d_in[7];
  const float* bv            = (const float*)d_in[8];
  const float* wo            = (const float*)d_in[9];
  const float* bo            = (const float*)d_in[10];
  const float* gamma         = (const float*)d_in[11];
  const float* beta          = (const float*)d_in[12];
  float* ws  = (float*)d_ws;
  float* out = (float*)d_out;

  const size_t shmem = (size_t)(24576 + 128 + 12288) * 4;  // ~144.5 KB
  (void)hipFuncSetAttribute((const void*)k_attn,
                            hipFuncAttributeMaxDynamicSharedMemorySize,
                            (int)shmem);

  k_proj_q<<<D_, 256, 0, stream>>>(query, wq, bq, ws);
  k_qk<<<48, 256, 0, stream>>>(wk, bk, ws);
  k_pack<<<KK_, 256, 0, stream>>>(ws);
  k_attn<<<B_ * CHUNKS_, 256, shmem, stream>>>(tokens, mask, ws);
  k_reduce<<<(B_ * H_ * D_) / 256, 256, 0, stream>>>(ws);
  k_wv<<<D_, 256, 0, stream>>>(wv, bv, ws);
  k_wo<<<D_, 256, 0, stream>>>(wo, bo, ws);
  k_ln<<<B_, 256, 0, stream>>>(gamma, beta, ws, out);
}
